// CTLN_55568286876127
// MI455X (gfx1250) — matrix operand order repaired and checked by reading, compiled, not run on hardware
//
#include <hip/hip_runtime.h>
#include <hip/hip_bf16.h>

// ---------------- problem constants ----------------
#define N_NEUR 1024
#define T_STEPS 16384
#define DT 0.1f

// ---------------- decomposition -------------------
#define NWG 16              // persistent workgroups (one W slice each)
#define ROWS_PER_WG 64      // N_NEUR / NWG
#define BLOCK 128           // 4 waves (wave32)
#define ROWS_PER_WAVE 16    // one 16x16 WMMA row-tile per wave
#define PADW 1032           // LDS row stride in f16 elems: 516 dwords, %64==4 -> conflict-free b128 reads

typedef _Float16 v8h  __attribute__((ext_vector_type(8)));
typedef _Float16 v16h __attribute__((ext_vector_type(16)));
typedef float    v8f  __attribute__((ext_vector_type(8)));

#define SMEM_BYTES (ROWS_PER_WG * PADW * 2 + N_NEUR * 2)

// ---------------------------------------------------------------------------
// Init kernel: reset grid-barrier state and seed x buffer with x0.
// Runs before the persistent kernel on the same stream each call, so the
// launch is deterministic under graph replay (no stale ws state).
// ---------------------------------------------------------------------------
__global__ void ctln_init(const float* __restrict__ x0,
                          float* __restrict__ xbuf,
                          unsigned* cnt, unsigned* phase) {
  int i = blockIdx.x * blockDim.x + threadIdx.x;
  if (i == 0) { *cnt = 0u; *phase = 0u; }
  if (i < N_NEUR) xbuf[i] = x0[i];
}

// ---------------------------------------------------------------------------
// Persistent CTLN kernel. Each WG owns 64 rows of W, held in LDS as f16 for
// the whole run. Per step: rebuild f16 x in LDS, run 32 chained
// v_wmma_f32_16x16x32_f16 per wave (K = 1024), pointwise Euler update in f32
// registers, publish x to the double-buffered global vector, grid barrier.
// ---------------------------------------------------------------------------
__launch_bounds__(BLOCK, 1)
__global__ void ctln_persistent(const float* __restrict__ u,
                                const float* __restrict__ W,
                                const float* __restrict__ theta,
                                float* __restrict__ out,
                                float* __restrict__ xbuf,   // [2][N_NEUR]
                                unsigned* cnt, unsigned* phase) {
  extern __shared__ char smem[];
  _Float16* Wl = (_Float16*)smem;                              // [ROWS_PER_WG][PADW]
  _Float16* xl = (_Float16*)(smem + ROWS_PER_WG * PADW * 2);   // [N_NEUR]

  const int wg   = blockIdx.x;
  const int tid  = threadIdx.x;
  const int lane = tid & 31;
  const int wave = tid >> 5;
  const int row0 = wg * ROWS_PER_WG;

  // ---- one-time: stage this WG's W slice into LDS as f16 (padded rows)
  for (int idx = tid; idx < ROWS_PER_WG * N_NEUR; idx += BLOCK) {
    int r = idx >> 10;          // / N_NEUR
    int c = idx & (N_NEUR - 1);
    Wl[r * PADW + c] = (_Float16)W[(size_t)(row0 + r) * N_NEUR + c];
  }

  // Pointwise-update ownership: lanes 0 and 16 each own 8 rows of the D tile
  // (D layout: vgpr r -> M=r for lanes 0-15 / M=8+r for lanes 16-31, N=lane%16).
  const bool updater   = (lane == 0) || (lane == 16);
  const int  mloc_base = wave * ROWS_PER_WAVE + ((lane == 16) ? 8 : 0);
  float xown[8];   // f32 state kept in registers for full-precision Euler steps
  float th[8];
  if (updater) {
    for (int r = 0; r < 8; ++r) {
      int mg = row0 + mloc_base + r;
      xown[r] = xbuf[mg];       // x0 seeded by init kernel
      th[r]   = theta[mg];
    }
  }
  __syncthreads();

  // ISA fragment maps (16-bit A 16x32, doc §7.12.2): lanes<16 -> K=base..base+7
  // and 16+base..16+base+7 with base=0; lanes>=16 -> base=8.
  const int base1 = (lane < 16) ? 0 : 8;    // A K offset
  const int base2 = (lane < 16) ? 0 : 8;    // B K offset (x broadcast columns)
  const int mlA   = wave * ROWS_PER_WAVE + (lane & 15);
  const _Float16* arow = Wl + mlA * PADW;

  for (int t = 0; t < T_STEPS; ++t) {
    const int cur = t & 1;
    const float* xcur = xbuf + cur * N_NEUR;
    float*       xnxt = xbuf + (cur ^ 1) * N_NEUR;

    // refresh f16 copy of x in LDS (4 KB read, L2-resident)
    for (int n = tid; n < N_NEUR; n += BLOCK)
      xl[n] = (_Float16)xcur[n];
    __syncthreads();

    // y = W_slice @ x  via chained WMMA (f32 accumulate)
    v8f acc = {};
    #pragma unroll 4
    for (int k0 = 0; k0 < N_NEUR; k0 += 32) {
      union { v16h v; v8h h[2]; } a, b;
      a.h[0] = *(const v8h*)(arow + k0 + base1);
      a.h[1] = *(const v8h*)(arow + k0 + 16 + base1);
      b.h[0] = *(const v8h*)(xl + k0 + base2);
      b.h[1] = *(const v8h*)(xl + k0 + 16 + base2);
      acc = __builtin_amdgcn_wmma_f32_16x16x32_f16(
                false, a.v, false, b.v, (short)0, acc, false, false);
    }

    // pointwise update: x' = x + dt*(relu(y + theta + u[:,t]) - x)
    if (updater) {
      #pragma unroll
      for (int r = 0; r < 8; ++r) {
        int   mg   = row0 + mloc_base + r;
        float pre  = acc[r] + th[r] + u[(size_t)mg * T_STEPS + t];
        float rl   = pre > 0.0f ? pre : 0.0f;
        float xn   = xown[r] + DT * (rl - xown[r]);
        xown[r]    = xn;
        xnxt[mg]   = xn;
        out[(size_t)mg * T_STEPS + t] = xn;
      }
    }

    // ---- one grid barrier per step (sense-reversal, double-buffered x) ----
    __syncthreads();                       // also protects xl WAR
    if (tid == 0) {
      __threadfence();                     // release x writes to device scope
      unsigned my   = __hip_atomic_load(phase, __ATOMIC_RELAXED,
                                        __HIP_MEMORY_SCOPE_AGENT);
      unsigned prev = __hip_atomic_fetch_add(cnt, 1u, __ATOMIC_ACQ_REL,
                                             __HIP_MEMORY_SCOPE_AGENT);
      if (prev == NWG - 1) {
        __hip_atomic_store(cnt, 0u, __ATOMIC_RELAXED, __HIP_MEMORY_SCOPE_AGENT);
        __hip_atomic_fetch_add(phase, 1u, __ATOMIC_RELEASE,
                               __HIP_MEMORY_SCOPE_AGENT);
      } else {
        while (__hip_atomic_load(phase, __ATOMIC_ACQUIRE,
                                 __HIP_MEMORY_SCOPE_AGENT) == my) {
          __builtin_amdgcn_s_sleep(2);
        }
      }
      __threadfence();                     // acquire: drop stale WGP$ lines
    }
    __syncthreads();
  }
}

// ---------------------------------------------------------------------------
// Host launch. Inputs (setup_inputs order): x0[N], u[N*T], W[N*N], theta[N].
// Output: xs[N][T] f32. Workspace layout: cnt @0, phase @128, xbuf[2][N] @256.
// ---------------------------------------------------------------------------
extern "C" void kernel_launch(void* const* d_in, const int* in_sizes, int n_in,
                              void* d_out, int out_size, void* d_ws, size_t ws_size,
                              hipStream_t stream) {
  const float* x0    = (const float*)d_in[0];
  const float* u     = (const float*)d_in[1];
  const float* W     = (const float*)d_in[2];
  const float* theta = (const float*)d_in[3];
  float* out = (float*)d_out;

  char* ws = (char*)d_ws;
  unsigned* cnt   = (unsigned*)(ws);
  unsigned* phase = (unsigned*)(ws + 128);
  float*    xbuf  = (float*)(ws + 256);

  ctln_init<<<(N_NEUR + 255) / 256, 256, 0, stream>>>(x0, xbuf, cnt, phase);

  (void)hipFuncSetAttribute(reinterpret_cast<const void*>(ctln_persistent),
                            hipFuncAttributeMaxDynamicSharedMemorySize,
                            (int)SMEM_BYTES);
  ctln_persistent<<<NWG, BLOCK, SMEM_BYTES, stream>>>(u, W, theta, out,
                                                      xbuf, cnt, phase);
}
